// Mamba_Block_63874753626312
// MI455X (gfx1250) — compile-verified, matrix operand order checked by reading, not run on hardware
//
#include <hip/hip_runtime.h>
#include <hip/hip_bf16.h>
#include <stdint.h>

typedef __attribute__((ext_vector_type(16))) _Float16 v16h;
typedef __attribute__((ext_vector_type(8)))  _Float16 v8h;
typedef __attribute__((ext_vector_type(8)))  float    v8f;
typedef __attribute__((ext_vector_type(4)))  float    v4f;
typedef __attribute__((ext_vector_type(4)))  unsigned int u32x4;
typedef __attribute__((ext_vector_type(8)))  unsigned int u32x8;

// 16-byte global -> LDS async copy (GLOBAL_LOAD_ASYNC_TO_LDS_B128, ASYNCcnt).
__device__ __forceinline__ void cp_async16(void* lds_dst, const void* gsrc) {
  uint32_t loff = (uint32_t)(uintptr_t)lds_dst;
  uint64_t g    = (uint64_t)(uintptr_t)gsrc;
  asm volatile("global_load_async_to_lds_b128 %0, %1, off" :: "v"(loff), "v"(g) : "memory");
}
template <int N>
__device__ __forceinline__ void wait_async_le() {
  if constexpr (N == 0)      asm volatile("s_wait_asynccnt 0x0" ::: "memory");
  else if constexpr (N == 1) asm volatile("s_wait_asynccnt 0x1" ::: "memory");
  else                       asm volatile("s_wait_asynccnt 0x2" ::: "memory");
}
template <int N>
__device__ __forceinline__ void wait_tensor_le() {
#if __has_builtin(__builtin_amdgcn_s_wait_tensorcnt)
  __builtin_amdgcn_s_wait_tensorcnt(N);  // immediate operand (SIMM16)
#else
  if constexpr (N == 0) asm volatile("s_wait_tensorcnt 0x0" ::: "memory");
  else                  asm volatile("s_wait_tensorcnt 0x1" ::: "memory");
#endif
}

// TDM: move a rows x 32-halves 2D tile from global (row stride ldElems halves)
// into LDS, padding 4 DWORDs after each 16-DWORD (64B) row -> LDS row stride
// 80B (= 40 halves, matching the compute-side layout). D# per ISA ch.8;
// groups 2/3 are NULL (2D tensor).
__device__ __forceinline__ void tdm_load_tile_b16(
    const _Float16* gsrc, void* lds_dst, int ldElems,
    int rows, int remRows, int remK) {
  uint64_t ga = (uint64_t)(uintptr_t)gsrc;
  u32x4 g0;
  g0[0] = 1u;                                                  // count=1
  g0[1] = (uint32_t)(uintptr_t)lds_dst;                        // lds_addr
  g0[2] = (uint32_t)ga;                                        // global_addr lo
  g0[3] = (uint32_t)((ga >> 32) & 0x1FFFFFFu) | (2u << 30);    // addr hi | type=2
  uint32_t td0 = (uint32_t)remK;      // tensor_dim0 (elements past tile start)
  uint32_t td1 = (uint32_t)remRows;   // tensor_dim1
  u32x8 g1;
  g1[0] = (1u << 16)      // data_size = 2 bytes
        | (1u << 20)      // pad_enable
        | (3u << 22)      // pad_interval: 16 DWORDs
        | (3u << 25);     // pad_amount:   4 DWORDs
  g1[1] = (td0 & 0xFFFFu) << 16;                   // tensor_dim0[15:0]
  g1[2] = (td0 >> 16) | ((td1 & 0xFFFFu) << 16);   // dim0 hi | dim1 lo
  g1[3] = (td1 >> 16) | (32u << 16);               // dim1 hi | tile_dim0=32
  g1[4] = (uint32_t)rows;                          // tile_dim1 (tile_dim2=0)
  g1[5] = (uint32_t)ldElems;                       // tensor_dim0_stride lo
  g1[6] = 0u;                                      // stride hi | dim1_stride lo
  g1[7] = 0u;
  asm volatile("tensor_load_to_lds %0, %1" :: "s"(g0), "s"(g1) : "memory");
}

// Fragment load for v_wmma_f32_16x16x32_f16 (two b128 LDS reads per lane).
__device__ __forceinline__ v16h load_frag(const _Float16* base, int lane) {
  int m = lane & 15, kh = lane >> 4;
  const _Float16* p = base + m * 40 + kh * 8;
  v8h lo = *(const v8h*)(p);
  v8h hi = *(const v8h*)(p + 16);
  v16h r;
#pragma unroll
  for (int i = 0; i < 8; ++i) { r[i] = lo[i]; r[i + 8] = hi[i]; }
  return r;
}

// C = A(MxK) * B(KxN), B pre-transposed to NxK f16 ("Bt", nbAlloc rows).
// EPI: 0 plain, 1 softplus(x+bias[n]), 2 resid[m,n]+x+bias[n].
// AF32: A is f32, converted while staging (B staged via per-lane async).
// !AF32: A staged via per-lane async, B tile staged via TDM (wave 0).
template <int EPI, bool AF32>
__global__ __launch_bounds__(256) void gemm_kernel(
    const void* Aany, int lda, const _Float16* Bt, int ldb, int nbAlloc,
    float* C, int ldc, int M, int N, int K,
    const float* bias, const float* resid) {
  constexpr int BM = 64, BN = 128, BK = 32, SA = 40;
  constexpr int NPS = AF32 ? 2 : 1;  // per-wave async instrs per stage
  __shared__ _Float16 LA[2][BM * SA];  // 2 x 5 KB
  __shared__ _Float16 LB[2][BN * SA];  // 2 x 10 KB
  const int tid = threadIdx.x;
  const int m0 = blockIdx.y * BM;
  const int n0 = blockIdx.x * BN;
  const int wid = tid >> 5, lane = tid & 31;
  const int wm = wid & 1, wn = wid >> 1;  // 2 x 4 wave grid, 32x32 per wave

  v8f acc[2][2] = {};

  const _Float16* A16 = (const _Float16*)Aany;
  const float*    A32 = (const float*)Aany;

  const int ar = tid >> 2, ac = (tid & 3) * 8;  // 64 rows x 4 chunks of 8 halves

  auto stage = [&](int buf, int kb) {
    if constexpr (AF32) {
      // B tile: 2 x 16B per thread, async
#pragma unroll
      for (int i = 0; i < 2; ++i) {
        int idx = tid + i * 256;
        int r = idx >> 2, c = (idx & 3) * 8;
        cp_async16(&LB[buf][r * SA + c], &Bt[(size_t)(n0 + r) * ldb + kb + c]);
      }
      // A tile: f32 -> f16 in-register, ds store
      const float* ap = &A32[(size_t)(m0 + ar) * lda + kb + ac];
      v4f f0 = *(const v4f*)(ap);
      v4f f1 = *(const v4f*)(ap + 4);
      _Float16* lp = &LA[buf][ar * SA + ac];
#pragma unroll
      for (int i = 0; i < 4; ++i) { lp[i] = (_Float16)f0[i]; lp[4 + i] = (_Float16)f1[i]; }
    } else {
      // B tile: one TDM descriptor, issued by wave 0 only (EXEC-independent DMA)
      if (wid == 0)
        tdm_load_tile_b16(&Bt[(size_t)n0 * ldb + kb], &LB[buf][0], ldb,
                          BN, nbAlloc - n0, K - kb);
      // A tile: per-lane async (1 instr per wave)
      cp_async16(&LA[buf][ar * SA + ac], &A16[(size_t)(m0 + ar) * lda + kb + ac]);
    }
  };

  stage(0, 0);  // prologue
  int cur = 0;
  for (int kb = 0; kb < K; kb += BK) {
    const bool has_next = (kb + BK) < K;
    if (has_next) {
      stage(cur ^ 1, kb + BK);                        // issue next stage
      __builtin_prefetch(&Bt[(size_t)(n0 + ar) * ldb + kb + 2 * BK], 0, 3);
      if (!AF32 && wid == 0) wait_tensor_le<1>();     // wait prev TDM only
      wait_async_le<NPS>();                           // wait prev async only
    } else {
      if (!AF32 && wid == 0) wait_tensor_le<0>();
      wait_async_le<0>();
    }
    __syncthreads();

    v16h a0 = load_frag(&LA[cur][(wm * 32 +  0) * SA], lane);
    v16h a1 = load_frag(&LA[cur][(wm * 32 + 16) * SA], lane);
    v16h b0 = load_frag(&LB[cur][(wn * 32 +  0) * SA], lane);
    v16h b1 = load_frag(&LB[cur][(wn * 32 + 16) * SA], lane);

    acc[0][0] = __builtin_amdgcn_wmma_f32_16x16x32_f16(false, a0, false, b0, (short)0, acc[0][0], false, false);
    acc[0][1] = __builtin_amdgcn_wmma_f32_16x16x32_f16(false, a0, false, b1, (short)0, acc[0][1], false, false);
    acc[1][0] = __builtin_amdgcn_wmma_f32_16x16x32_f16(false, a1, false, b0, (short)0, acc[1][0], false, false);
    acc[1][1] = __builtin_amdgcn_wmma_f32_16x16x32_f16(false, a1, false, b1, (short)0, acc[1][1], false, false);

    __syncthreads();  // all waves done reading buf cur before it is re-staged
    cur ^= 1;
  }

  // C/D layout: lanes 0-15: M=v, N=lane; lanes 16-31: M=8+v, N=lane-16.
  const int nl = lane & 15, mh = lane >> 4;
#pragma unroll
  for (int i = 0; i < 2; ++i)
#pragma unroll
    for (int j = 0; j < 2; ++j)
#pragma unroll
      for (int v = 0; v < 8; ++v) {
        int m = m0 + wm * 32 + i * 16 + mh * 8 + v;
        int n = n0 + wn * 32 + j * 16 + nl;
        float val = acc[i][j][v];
        if (EPI == 1) { val += bias[n]; val = (val > 20.f) ? val : __logf(1.f + __expf(val)); }
        if (EPI == 2) { val += bias[n] + resid[(size_t)m * ldc + n]; }
        if (n < N) C[(size_t)m * ldc + n] = val;
      }
}

// mean-pool (k=2) + RMSNorm; writes f32 (residual) and f16 (GEMM A) copies.
__global__ __launch_bounds__(256) void pool_rmsnorm(
    const float* __restrict__ x, const float* __restrict__ w,
    float* __restrict__ xn, _Float16* __restrict__ xnh) {
  int t = blockIdx.x;            // token 0..16383
  int b = t >> 11, l = t & 2047;
  const float* r0 = x + ((size_t)b * 4096 + 2 * l) * 1024;
  const float* r1 = r0 + 1024;
  float v[4], ss = 0.f;
#pragma unroll
  for (int i = 0; i < 4; ++i) {
    int c = threadIdx.x + i * 256;
    float a = 0.5f * (r0[c] + r1[c]);
    v[i] = a; ss += a * a;
  }
  __shared__ float red[256];
  red[threadIdx.x] = ss;
  __syncthreads();
  for (int s = 128; s > 0; s >>= 1) {
    if (threadIdx.x < s) red[threadIdx.x] += red[threadIdx.x + s];
    __syncthreads();
  }
  float scale = rsqrtf(red[0] * (1.f / 1024.f) + 1e-5f);
#pragma unroll
  for (int i = 0; i < 4; ++i) {
    int c = threadIdx.x + i * 256;
    float o = v[i] * scale * w[c];
    xn[(size_t)t * 1024 + c] = o;
    xnh[(size_t)t * 1024 + c] = (_Float16)o;
  }
}

// Wt[n*K + k] = f16(W[k*N + n]); rows n in [N, Npad) zero-padded.
__global__ __launch_bounds__(256) void transpose_cvt(
    const float* __restrict__ W, _Float16* __restrict__ Wt, int K, int N, int Npad) {
  size_t idx = (size_t)blockIdx.x * 256 + threadIdx.x;
  if (idx >= (size_t)Npad * K) return;
  int n = (int)(idx / K), k = (int)(idx % K);
  float v = (n < N) ? W[(size_t)k * N + n] : 0.f;
  Wt[idx] = (_Float16)v;
}

// depthwise causal conv (4 taps along L) + SiLU on the xm half of xz.
__global__ __launch_bounds__(256) void conv_silu(
    const float* __restrict__ xz, const float* __restrict__ cw,
    const float* __restrict__ cb, _Float16* __restrict__ xch) {
  size_t idx = (size_t)blockIdx.x * 256 + threadIdx.x;  // over T*2048
  if (idx >= (size_t)16384 * 2048) return;
  int d = (int)(idx & 2047);
  long t = (long)(idx >> 11);
  int l = (int)(t & 2047);
  float acc = cb[d];
#pragma unroll
  for (int k = 0; k < 4; ++k) {
    int off = k - 3;
    if (l + off >= 0) acc += cw[d * 4 + k] * xz[(size_t)(t + off) * 4096 + d];
  }
  float s = acc * (1.f / (1.f + __expf(-acc)));  // silu
  xch[idx] = (_Float16)s;
}

// selective scan: one lane per (b,d) channel, 16 states in registers,
// B_t/C_t double-buffered through LDS (one barrier per timestep).
__global__ __launch_bounds__(256) void scan_kernel(
    const float* __restrict__ dt, const _Float16* __restrict__ xch,
    const float* __restrict__ xdbl, const float* __restrict__ A_log,
    float* __restrict__ y) {
  int b = blockIdx.x >> 3;
  int d = (blockIdx.x & 7) * 256 + threadIdx.x;  // 0..2047
  float A[16], h[16];
#pragma unroll
  for (int s = 0; s < 16; ++s) { A[s] = -__expf(A_log[d * 16 + s]); h[s] = 0.f; }
  __shared__ float BC[2][32];
  size_t base = (size_t)b * 2048;
  if (threadIdx.x < 32) BC[0][threadIdx.x] = xdbl[base * 96 + 64 + threadIdx.x];
  __syncthreads();
  for (int l = 0; l < 2048; ++l) {
    int cur = l & 1;
    if (l + 1 < 2048 && threadIdx.x < 32)
      BC[cur ^ 1][threadIdx.x] = xdbl[(base + l + 1) * 96 + 64 + threadIdx.x];
    size_t t = base + l;
    float dtv = dt[t * 2048 + d];
    float xv  = (float)xch[t * 2048 + d];
    float dtx = dtv * xv;
    float yv = 0.f;
#pragma unroll
    for (int s = 0; s < 16; ++s) {
      h[s] = __expf(dtv * A[s]) * h[s] + dtx * BC[cur][s];
      yv += h[s] * BC[cur][16 + s];
    }
    y[t * 2048 + d] = yv;
    __syncthreads();
  }
}

// y = (y + xc*D) * silu(z) -> f16
__global__ __launch_bounds__(256) void gate_kernel(
    const float* __restrict__ y, const _Float16* __restrict__ xch,
    const float* __restrict__ xz, const float* __restrict__ Dp,
    _Float16* __restrict__ yh) {
  size_t idx = (size_t)blockIdx.x * 256 + threadIdx.x;
  if (idx >= (size_t)16384 * 2048) return;
  int d = (int)(idx & 2047);
  size_t t = idx >> 11;
  float z = xz[t * 4096 + 2048 + d];
  float g = z * (1.f / (1.f + __expf(-z)));
  float v = (y[idx] + (float)xch[idx] * Dp[d]) * g;
  yh[idx] = (_Float16)v;
}

extern "C" void kernel_launch(void* const* d_in, const int* in_sizes, int n_in,
                              void* d_out, int out_size, void* d_ws, size_t ws_size,
                              hipStream_t stream) {
  (void)in_sizes; (void)n_in; (void)out_size; (void)ws_size;
  const float* x        = (const float*)d_in[0];
  const float* norm1_w  = (const float*)d_in[1];
  const float* lin1_w   = (const float*)d_in[2];
  const float* lin1_b   = (const float*)d_in[3];
  const float* in_proj  = (const float*)d_in[4];
  const float* conv_w   = (const float*)d_in[5];
  const float* conv_b   = (const float*)d_in[6];
  const float* x_proj   = (const float*)d_in[7];
  const float* dt_proj  = (const float*)d_in[8];
  const float* dt_b     = (const float*)d_in[9];
  const float* A_log    = (const float*)d_in[10];
  const float* Dp       = (const float*)d_in[11];
  const float* out_proj = (const float*)d_in[12];
  float* out = (float*)d_out;

  char* w = (char*)d_ws;
  size_t off = 0;
  auto alloc = [&](size_t bytes) {
    void* p = w + off;
    off = (off + bytes + 255) & ~(size_t)255;
    return p;
  };
  const size_t T = 16384;
  float*    XN   = (float*)   alloc(T * 1024 * 4);          // rmsnorm (residual)
  _Float16* XNH  = (_Float16*)alloc(T * 1024 * 2);
  _Float16* WINT = (_Float16*)alloc((size_t)4096 * 1024 * 2); // in_proj^T f16
  float*    XZ   = (float*)   alloc(T * 4096 * 4);          // [xm | z]
  _Float16* XCH  = (_Float16*)alloc(T * 2048 * 2);          // conv+silu f16
  _Float16* WXPT = (_Float16*)alloc((size_t)128 * 2048 * 2); // x_proj^T, N padded to 128
  float*    XDBL = (float*)   alloc(T * 96 * 4);            // [dt(64) | B(16) | C(16)]
  _Float16* WDTT = (_Float16*)alloc((size_t)2048 * 64 * 2); // dt_proj^T
  float*    DTB  = (float*)   alloc(T * 2048 * 4);          // softplus(dt)
  float*    YB   = (float*)   alloc(T * 2048 * 4);          // scan out
  _Float16* YH   = (_Float16*)alloc(T * 2048 * 2);          // gated y f16
  _Float16* L1T  = (_Float16*)alloc((size_t)1024 * 1024 * 2); // lin1^T
  float*    WF   = (float*)   alloc((size_t)2048 * 1024 * 4); // out_proj@lin1 f32
  _Float16* WFT  = (_Float16*)alloc((size_t)1024 * 2048 * 2); // fused^T f16

  pool_rmsnorm<<<16384, 256, 0, stream>>>(x, norm1_w, XN, XNH);

  auto tc = [&](const float* W, _Float16* Wt, int K, int N, int Npad) {
    size_t tot = (size_t)Npad * K;
    transpose_cvt<<<(unsigned)((tot + 255) / 256), 256, 0, stream>>>(W, Wt, K, N, Npad);
  };
  tc(in_proj, WINT, 1024, 4096, 4096);
  tc(x_proj,  WXPT, 2048, 96, 128);
  tc(dt_proj, WDTT, 64, 2048, 2048);
  tc(lin1_w,  L1T,  1024, 1024, 1024);

  // xz = xn @ in_proj  (16384 x 1024 x 4096)  [B tiles via TDM]
  gemm_kernel<0, false><<<dim3(4096 / 128, 16384 / 64), 256, 0, stream>>>(
      XNH, 1024, WINT, 1024, 4096, XZ, 4096, 16384, 4096, 1024, nullptr, nullptr);

  conv_silu<<<(unsigned)((T * 2048 + 255) / 256), 256, 0, stream>>>(XZ, conv_w, conv_b, XCH);

  // x_dbl = xc @ x_proj  (16384 x 2048 x 96)
  gemm_kernel<0, false><<<dim3(1, 16384 / 64), 256, 0, stream>>>(
      XCH, 2048, WXPT, 2048, 128, XDBL, 96, 16384, 96, 2048, nullptr, nullptr);

  // dt = softplus(x_dbl[:, :64] @ dt_proj + dt_b)  (16384 x 64 x 2048)
  gemm_kernel<1, true><<<dim3(2048 / 128, 16384 / 64), 256, 0, stream>>>(
      XDBL, 96, WDTT, 64, 2048, DTB, 2048, 16384, 2048, 64, dt_b, nullptr);

  // Wfused = out_proj @ lin1  (2048 x 1024 x 1024)
  gemm_kernel<0, true><<<dim3(1024 / 128, 2048 / 64), 256, 0, stream>>>(
      out_proj, 1024, L1T, 1024, 1024, WF, 1024, 2048, 1024, 1024, nullptr, nullptr);
  tc(WF, WFT, 2048, 1024, 1024);

  scan_kernel<<<64, 256, 0, stream>>>(DTB, XCH, XDBL, A_log, YB);
  gate_kernel<<<(unsigned)((T * 2048 + 255) / 256), 256, 0, stream>>>(YB, XCH, XZ, Dp, YH);

  // out = xn + y @ Wfused + lin1_b  (16384 x 2048 x 1024)  [B tiles via TDM]
  gemm_kernel<2, false><<<dim3(1024 / 128, 16384 / 64), 256, 0, stream>>>(
      YH, 2048, WFT, 2048, 1024, out, 1024, 16384, 1024, 2048, lin1_b, XN);
}
